// Contrast_30039001269017
// MI455X (gfx1250) — matrix operand order repaired and checked by reading, not compiled, not run on hardware
//
#include <hip/hip_runtime.h>
#include <hip/hip_bf16.h>

// ---------------------------------------------------------------------------
// Contrastive loss for MI455X (gfx1250), wave32 + bf16 WMMA + async-to-LDS.
//
//   p1 = elu(z_mp @ W1^T + b1) @ W2^T + b2 ; p2 likewise   (4 GEMMs, 17 GF)
//   n1, n2 = row-normalized p1, p2
//   S = exp(n1 @ n2^T / tau)   -> fused row/col sums, never materialized
//   per-edge S entries recomputed directly (cheap)
//   loss = 0.5*mean(-log s_mp) + 0.5*mean(-log s_sc)
//
// sim_kernel double-buffers its LDS tiles with GLOBAL_LOAD_ASYNC_TO_LDS_B128
// (ASYNCcnt) so the next K-chunk's DMA overlaps the current chunk's WMMAs.
// ---------------------------------------------------------------------------

static constexpr int kN   = 8192;
static constexpr int kH   = 512;
static constexpr float kInvTau = 1.25f;   // 1 / 0.8

typedef __attribute__((ext_vector_type(16))) __bf16          v16bf;
typedef __attribute__((ext_vector_type(8)))  float           v8f;
typedef __attribute__((ext_vector_type(8)))  unsigned short  u16x8;
typedef __attribute__((ext_vector_type(16))) unsigned short  u16x16;
typedef __attribute__((ext_vector_type(4)))  float           f32x4;

__device__ __forceinline__ unsigned short f2bf(float f) {
  unsigned int u = __builtin_bit_cast(unsigned int, f);
  u += 0x7FFFu + ((u >> 16) & 1u);          // round-to-nearest-even
  return (unsigned short)(u >> 16);
}
__device__ __forceinline__ float bf2f(unsigned short h) {
  unsigned int u = ((unsigned int)h) << 16;
  return __builtin_bit_cast(float, u);
}
__device__ __forceinline__ v16bf mk_frag(u16x8 lo, u16x8 hi) {
  u16x16 t;
#pragma unroll
  for (int i = 0; i < 8; ++i) { t[i] = lo[i]; t[i + 8] = hi[i]; }
  return __builtin_bit_cast(v16bf, t);
}

// ---------------------------------------------------------------------------
// Prep: convert W1/W2 (512x512 f32) to bf16, zero global accumulators.
// ---------------------------------------------------------------------------
__global__ void prep_kernel(const float* __restrict__ W1,
                            const float* __restrict__ W2,
                            unsigned short* __restrict__ W1b,
                            unsigned short* __restrict__ W2b,
                            float* __restrict__ rowsum, float* __restrict__ colsum,
                            float* __restrict__ s_mp,   float* __restrict__ s_sc) {
  int i = blockIdx.x * blockDim.x + threadIdx.x;    // 0..262143
  W1b[i] = f2bf(W1[i]);
  W2b[i] = f2bf(W2[i]);
  if (i < kN) { rowsum[i] = 0.f; colsum[i] = 0.f; s_mp[i] = 0.f; s_sc[i] = 0.f; }
}

// ---------------------------------------------------------------------------
// Fused projection + normalize. One block = 64 rows of one input.
// LDS: Zs (64x520 bf16) -> Hs (64x520 bf16) -> P reuses Zs; rnorm[64].
// 8 waves: wave w -> row-tile (w>>1), column half (w&1).
// Each wave processes 4 col-tiles per A-fragment: 4 independent accumulator
// chains per K-step so frag-load latency hides behind WMMA issue.
// ---------------------------------------------------------------------------
static constexpr int kStr1 = kH + 8;   // 520: padded LDS k-stride, keeps 16B align

__global__ void proj_norm_kernel(const float* __restrict__ z_mp,
                                 const float* __restrict__ z_sc,
                                 const unsigned short* __restrict__ W1b,
                                 const float* __restrict__ bias1,
                                 const unsigned short* __restrict__ W2b,
                                 const float* __restrict__ bias2,
                                 unsigned short* __restrict__ n1,
                                 unsigned short* __restrict__ n2) {
  extern __shared__ char smem[];
  unsigned short* Zs = (unsigned short*)smem;          // 64 x kStr1 (reused as P)
  unsigned short* Hs = Zs + 64 * kStr1;                // 64 x kStr1
  float* rnorm = (float*)(Hs + 64 * kStr1);            // 64

  const float* z = (blockIdx.y == 0) ? z_mp : z_sc;
  unsigned short* nout = (blockIdx.y == 0) ? n1 : n2;
  const int row0 = blockIdx.x * 64;
  const int t = threadIdx.x;
  const int w = t >> 5, lane = t & 31;
  const int hf = lane >> 4, ln = lane & 15;
  const int rt = w >> 1, ch = w & 1;
  const int mrow = rt * 16 + ln;

  if (t < 64) rnorm[t] = 0.0f;

  // stage Z block (f32 -> bf16) into LDS
#pragma unroll
  for (int i = 0; i < 32; ++i) {
    int seg = i * 256 + t;                 // 8192 segments of 4 floats
    int row = seg >> 7;                    // 128 segs per row
    int k   = (seg & 127) * 4;
    f32x4 zv = *(const f32x4*)(z + (size_t)(row0 + row) * kH + k);
    unsigned short* dst = Zs + row * kStr1 + k;
    dst[0] = f2bf(zv.x); dst[1] = f2bf(zv.y); dst[2] = f2bf(zv.z); dst[3] = f2bf(zv.w);
  }
  __syncthreads();

  v8f zero = {};

  // ---- Stage 1: H = elu(Z @ W1^T + b1) ----
  for (int g = 0; g < 4; ++g) {
    const int colg = ch * 256 + g * 64;
    v8f acc[4];
#pragma unroll
    for (int c = 0; c < 4; ++c) acc[c] = zero;
    const unsigned short* arow  = Zs + mrow * kStr1 + hf * 8;
    const unsigned short* brow0 = W1b + (size_t)(colg + ln) * kH + hf * 8;
#pragma unroll
    for (int k0 = 0; k0 < kH; k0 += 32) {
      v16bf af = mk_frag(*(const u16x8*)(arow + k0), *(const u16x8*)(arow + k0 + 16));
#pragma unroll
      for (int c = 0; c < 4; ++c) {
        const unsigned short* br = brow0 + (size_t)c * 16 * kH + k0;
        v16bf bf = mk_frag(*(const u16x8*)br, *(const u16x8*)(br + 16));
        acc[c] = __builtin_amdgcn_wmma_f32_16x16x32_bf16(false, af, false, bf,
                                                         (short)0, acc[c], false, false);
      }
    }
#pragma unroll
    for (int c = 0; c < 4; ++c) {
      const int col = colg + c * 16 + ln;
      const float bb = bias1[col];
#pragma unroll
      for (int v = 0; v < 8; ++v) {
        int m = rt * 16 + v + 8 * hf;
        float x = acc[c][v] + bb;
        float h = (x > 0.0f) ? x : (__expf(x) - 1.0f);   // ELU
        Hs[m * kStr1 + col] = f2bf(h);
      }
    }
  }
  __syncthreads();

  // ---- Stage 2: P = H @ W2^T + b2 ; accumulate row sum-of-squares ----
  unsigned short* Ps = Zs;   // Z is dead, reuse
  for (int g = 0; g < 4; ++g) {
    const int colg = ch * 256 + g * 64;
    v8f acc[4];
#pragma unroll
    for (int c = 0; c < 4; ++c) acc[c] = zero;
    const unsigned short* arow  = Hs + mrow * kStr1 + hf * 8;
    const unsigned short* brow0 = W2b + (size_t)(colg + ln) * kH + hf * 8;
#pragma unroll
    for (int k0 = 0; k0 < kH; k0 += 32) {
      v16bf af = mk_frag(*(const u16x8*)(arow + k0), *(const u16x8*)(arow + k0 + 16));
#pragma unroll
      for (int c = 0; c < 4; ++c) {
        const unsigned short* br = brow0 + (size_t)c * 16 * kH + k0;
        v16bf bf = mk_frag(*(const u16x8*)br, *(const u16x8*)(br + 16));
        acc[c] = __builtin_amdgcn_wmma_f32_16x16x32_bf16(false, af, false, bf,
                                                         (short)0, acc[c], false, false);
      }
    }
#pragma unroll
    for (int c = 0; c < 4; ++c) {
      const int col = colg + c * 16 + ln;
      const float bb = bias2[col];
#pragma unroll
      for (int v = 0; v < 8; ++v) {
        int m = rt * 16 + v + 8 * hf;
        float x = acc[c][v] + bb;
        Ps[m * kStr1 + col] = f2bf(x);
        atomicAdd(&rnorm[m], x * x);                     // ds_add_f32
      }
    }
  }
  __syncthreads();

  // ---- normalize rows, emit bf16 n ----
  {
    const int r  = t >> 2;
    const int c0 = (t & 3) * 128;
    const float sc = rsqrtf(rnorm[r] + 1e-30f);
    const unsigned short* src = Ps + r * kStr1 + c0;
    unsigned short* dst = nout + (size_t)(row0 + r) * kH + c0;
#pragma unroll 16
    for (int k = 0; k < 128; ++k) dst[k] = f2bf(bf2f(src[k]) * sc);
  }
}

// ---------------------------------------------------------------------------
// Similarity: per block, a 128x128 tile of S = exp(n1 @ n2^T / tau).
// Double-buffered LDS tiles filled by GLOBAL_LOAD_ASYNC_TO_LDS_B128; the
// next K-chunk's DMA overlaps the current chunk's WMMAs (s_wait_asynccnt 8).
// Fused exp + row/col partial sums -> global atomics. S never hits memory.
// 8 waves as 2 (row bands) x 4 (col bands); each wave: 4x2 16x16 tiles.
// ---------------------------------------------------------------------------
static constexpr int kStr2  = 72;            // padded k-stride (144 B, 16B aligned)
static constexpr int kTileE = 128 * kStr2;   // elems per 128x64 staged tile

__global__ void sim_kernel(const unsigned short* __restrict__ n1,
                           const unsigned short* __restrict__ n2,
                           float* __restrict__ rowsum,
                           float* __restrict__ colsum) {
  extern __shared__ char smem2[];
  unsigned short* Ab = (unsigned short*)smem2;       // 2 x (128 x 72)
  unsigned short* Bb = Ab + 2 * kTileE;              // 2 x (128 x 72)
  float* rsum = (float*)(Bb + 2 * kTileE);           // 128
  float* csum = rsum + 128;                          // 128

  const int t = threadIdx.x;
  const int w = t >> 5, lane = t & 31;
  const int hf = lane >> 4, ln = lane & 15;
  const int wr = w >> 2, wc = w & 3;
  const int row0 = blockIdx.y * 128;
  const int col0 = blockIdx.x * 128;

  if (t < 128) { rsum[t] = 0.0f; csum[t] = 0.0f; }

  // async-stage one 128x64 chunk of A (n1 rows) and B (n2 rows) into buffer
  // `buf`: 8 GLOBAL_LOAD_ASYNC_TO_LDS_B128 per wave per call (ASYNCcnt += 8).
  auto stage = [&](int buf, int k0) {
#pragma unroll
    for (int i = 0; i < 4; ++i) {
      int seg = i * 256 + t;               // 1024 segments of 8 elems
      int row = seg >> 3;
      int kk  = (seg & 7) * 8;
      unsigned lda = (unsigned)(size_t)(Ab + buf * kTileE + row * kStr2 + kk);
      unsigned ldb = (unsigned)(size_t)(Bb + buf * kTileE + row * kStr2 + kk);
      unsigned long long ga =
          (unsigned long long)(size_t)(n1 + (size_t)(row0 + row) * kH + k0 + kk);
      unsigned long long gb =
          (unsigned long long)(size_t)(n2 + (size_t)(col0 + row) * kH + k0 + kk);
      asm volatile("global_load_async_to_lds_b128 %0, %1, off"
                   :: "v"(lda), "v"(ga) : "memory");
      asm volatile("global_load_async_to_lds_b128 %0, %1, off"
                   :: "v"(ldb), "v"(gb) : "memory");
    }
  };

  v8f zero = {};
  v8f acc[4][2];
#pragma unroll
  for (int i = 0; i < 4; ++i)
#pragma unroll
    for (int j = 0; j < 2; ++j) acc[i][j] = zero;

  stage(0, 0);   // prologue: fill buffer 0

  for (int k0 = 0; k0 < kH; k0 += 64) {
    const int cur = (k0 >> 6) & 1;
    if (k0 + 64 < kH) {
      stage(cur ^ 1, k0 + 64);             // prefetch next chunk into other buf
      asm volatile("s_wait_asynccnt 0x8" ::: "memory");   // cur's 8 loads done
    } else {
      asm volatile("s_wait_asynccnt 0x0" ::: "memory");   // last chunk done
    }
    __syncthreads();                       // publish LDS to all waves

    const unsigned short* As = Ab + cur * kTileE;
    const unsigned short* Bs = Bb + cur * kTileE;
#pragma unroll
    for (int kk0 = 0; kk0 < 64; kk0 += 32) {
      v16bf afr[4];
#pragma unroll
      for (int r = 0; r < 4; ++r) {
        const unsigned short* ar = As + (wr * 64 + r * 16 + ln) * kStr2 + kk0 + hf * 8;
        afr[r] = mk_frag(*(const u16x8*)ar, *(const u16x8*)(ar + 16));
      }
#pragma unroll
      for (int c = 0; c < 2; ++c) {
        const unsigned short* br = Bs + (wc * 32 + c * 16 + ln) * kStr2 + kk0 + hf * 8;
        v16bf bfr = mk_frag(*(const u16x8*)br, *(const u16x8*)(br + 16));
#pragma unroll
        for (int r = 0; r < 4; ++r)
          acc[r][c] = __builtin_amdgcn_wmma_f32_16x16x32_bf16(false, afr[r], false, bfr,
                                                              (short)0, acc[r][c], false, false);
      }
    }
    __syncthreads();                       // tile consumed; buf may be refilled
  }

  // epilogue: exp + fused row/col reductions
  float colacc[2] = {0.f, 0.f};
#pragma unroll
  for (int r = 0; r < 4; ++r) {
#pragma unroll
    for (int v = 0; v < 8; ++v) {
      float s0 = __expf(acc[r][0][v] * kInvTau);
      float s1 = __expf(acc[r][1][v] * kInvTau);
      colacc[0] += s0; colacc[1] += s1;
      float rv = s0 + s1;
      rv += __shfl_xor(rv, 1, 32);
      rv += __shfl_xor(rv, 2, 32);
      rv += __shfl_xor(rv, 4, 32);
      rv += __shfl_xor(rv, 8, 32);
      if (ln == 0) atomicAdd(&rsum[wr * 64 + r * 16 + v + 8 * hf], rv);
    }
  }
#pragma unroll
  for (int c = 0; c < 2; ++c) {
    float cv = colacc[c] + __shfl_xor(colacc[c], 16, 32);
    if (hf == 0) atomicAdd(&csum[wc * 32 + c * 16 + ln], cv);
  }
  __syncthreads();
  if (t < 128) {
    atomicAdd(&rowsum[row0 + t], rsum[t]);
    atomicAdd(&colsum[col0 + t], csum[t]);
  }
}

// ---------------------------------------------------------------------------
// Edges: one wave per edge; recompute the two S entries and scatter.
// ---------------------------------------------------------------------------
__global__ void edge_kernel(const unsigned short* __restrict__ n1,
                            const unsigned short* __restrict__ n2,
                            const float* __restrict__ rowsum,
                            const float* __restrict__ colsum,
                            const int* __restrict__ pos, int E,
                            float* __restrict__ s_mp, float* __restrict__ s_sc) {
  int e = blockIdx.x * 8 + (threadIdx.x >> 5);
  int lane = threadIdx.x & 31;
  if (e >= E) return;
  int r = pos[e];
  int c = pos[E + e];
  const unsigned short* a1 = n1 + (size_t)r * kH;   // n1[r]
  const unsigned short* a2 = n2 + (size_t)c * kH;   // n2[c]
  const unsigned short* c1 = n1 + (size_t)c * kH;   // n1[c]
  const unsigned short* c2 = n2 + (size_t)r * kH;   // n2[r]
  int o = lane * 16;
  float d1 = 0.f, d2 = 0.f;
#pragma unroll
  for (int i = 0; i < 16; ++i) {
    d1 += bf2f(a1[o + i]) * bf2f(a2[o + i]);
    d2 += bf2f(c1[o + i]) * bf2f(c2[o + i]);
  }
#pragma unroll
  for (int m = 1; m < 32; m <<= 1) {
    d1 += __shfl_xor(d1, m, 32);
    d2 += __shfl_xor(d2, m, 32);
  }
  if (lane == 0) {
    atomicAdd(&s_mp[r], __expf(d1 * kInvTau) / rowsum[r]);   // S[r,c]/rowsum[r]
    atomicAdd(&s_sc[r], __expf(d2 * kInvTau) / colsum[r]);   // S[c,r]/colsum[r]
  }
}

// ---------------------------------------------------------------------------
// Finalize: loss = mean( 0.5*(-log s_mp) + 0.5*(-log s_sc) )
// ---------------------------------------------------------------------------
__global__ void finalize_kernel(const float* __restrict__ s_mp,
                                const float* __restrict__ s_sc,
                                float* __restrict__ out) {
  __shared__ float red[256];
  int t = threadIdx.x;
  float acc = 0.f;
  for (int i = t; i < kN; i += 256)
    acc += 0.5f * (-__logf(s_mp[i])) + 0.5f * (-__logf(s_sc[i]));
  red[t] = acc;
  __syncthreads();
  for (int s = 128; s > 0; s >>= 1) {
    if (t < s) red[t] += red[t + s];
    __syncthreads();
  }
  if (t == 0) out[0] = red[0] / (float)kN;
}

// ---------------------------------------------------------------------------
extern "C" void kernel_launch(void* const* d_in, const int* in_sizes, int n_in,
                              void* d_out, int out_size, void* d_ws, size_t ws_size,
                              hipStream_t stream) {
  const float* z_mp = (const float*)d_in[0];
  const float* z_sc = (const float*)d_in[1];
  const float* W1   = (const float*)d_in[2];
  const float* b1   = (const float*)d_in[3];
  const float* W2   = (const float*)d_in[4];
  const float* b2   = (const float*)d_in[5];
  const int*   pos  = (const int*)d_in[6];
  const int E = in_sizes[6] / 2;

  // workspace layout (~17.9 MB)
  char* ws = (char*)d_ws;
  unsigned short* W1b = (unsigned short*)ws;  ws += (size_t)kH * kH * 2;
  unsigned short* W2b = (unsigned short*)ws;  ws += (size_t)kH * kH * 2;
  unsigned short* n1  = (unsigned short*)ws;  ws += (size_t)kN * kH * 2;
  unsigned short* n2  = (unsigned short*)ws;  ws += (size_t)kN * kH * 2;
  float* rowsum = (float*)ws;  ws += (size_t)kN * 4;
  float* colsum = (float*)ws;  ws += (size_t)kN * 4;
  float* s_mp   = (float*)ws;  ws += (size_t)kN * 4;
  float* s_sc   = (float*)ws;  ws += (size_t)kN * 4;

  prep_kernel<<<(kH * kH) / 256, 256, 0, stream>>>(W1, W2, W1b, W2b,
                                                   rowsum, colsum, s_mp, s_sc);

  size_t lds1 = (size_t)(2 * 64 * kStr1) * sizeof(unsigned short) + 64 * sizeof(float);
  proj_norm_kernel<<<dim3(kN / 64, 2), 256, lds1, stream>>>(z_mp, z_sc, W1b, b1,
                                                            W2b, b2, n1, n2);

  size_t lds2 = (size_t)(4 * kTileE) * sizeof(unsigned short) + 256 * sizeof(float);
  sim_kernel<<<dim3(kN / 128, kN / 128), 256, lds2, stream>>>(n1, n2, rowsum, colsum);

  edge_kernel<<<(E + 7) / 8, 256, 0, stream>>>(n1, n2, rowsum, colsum, pos, E, s_mp, s_sc);

  finalize_kernel<<<1, 256, 0, stream>>>(s_mp, s_sc, (float*)d_out);
}
